// MLPBlock_26036091749042
// MI455X (gfx1250) — 8-bit matrix operand order repaired and checked by reading, compiled, not run on hardware
//
#include <hip/hip_runtime.h>
#include <math.h>
#include <stdint.h>

typedef int v8i __attribute__((ext_vector_type(8)));


__global__ void init_slots_kernel(unsigned* slots) {
    if (threadIdx.x < 4) slots[threadIdx.x] = 0u;
}

__global__ void absmax_kernel(const float* __restrict__ src, int n,
                              unsigned* __restrict__ slot) {
    __shared__ unsigned smax;
    if (threadIdx.x == 0) smax = 0u;
    __syncthreads();
    float m = 0.0f;
    for (int i = blockIdx.x * blockDim.x + threadIdx.x; i < n;
         i += gridDim.x * blockDim.x)
        m = fmaxf(m, fabsf(src[i]));
    atomicMax(&smax, __float_as_uint(m));
    __syncthreads();
    if (threadIdx.x == 0) atomicMax(slot, smax);
}

__global__ void quant_pack_kernel(const float* __restrict__ src, int K, int nseg,
                                  const unsigned* __restrict__ slot,
                                  signed char* __restrict__ dst,
                                  float* __restrict__ corr) {
    __shared__ int andbuf[256];
    const int row = blockIdx.x;
    const int tid = threadIdx.x;
    const float scale = fmaxf(__uint_as_float(*slot) / 7.0f, 1e-8f);
    const float inv = 1.0f / scale;
    for (int s = 0; s < nseg; ++s) {
        const int k = s * 256 + tid;
        float q = rintf(src[(size_t)row * K + k] * inv);
        q = fminf(fmaxf(q, -8.0f), 7.0f);
        const int qi = (int)q;
        dst[(size_t)row * K + k] = (signed char)qi;
        andbuf[tid] = qi & 0xF;
        __syncthreads();
        for (int off = 128; off > 0; off >>= 1) {
            if (tid < off) andbuf[tid] &= andbuf[tid + off];
            __syncthreads();
        }
        if (tid == 0) {
            const int nib = andbuf[0];
            corr[row * nseg + s] = (float)((nib ^ 8) - 8);
        }
        __syncthreads();
    }
}

__device__ __forceinline__ v8i load_a_frag(const signed char* p) {
    const int* a32 = (const int*)p;
    v8i a;
    a[0] = a32[0];  a[1] = a32[1];  a[2] = a32[2];  a[3] = a32[3];
    a[4] = a32[8];  a[5] = a32[9];  a[6] = a32[10]; a[7] = a32[11];
    return a;
}
__device__ __forceinline__ v8i load_b_frag(const signed char* p) {
    const int4* b128 = (const int4*)p;
    int4 t0 = b128[0], t1 = b128[2];
    v8i b;
    b[0] = t0.x; b[1] = t0.y; b[2] = t0.z; b[3] = t0.w;
    b[4] = t1.x; b[5] = t1.y; b[6] = t1.z; b[7] = t1.w;
    return b;
}

template <bool DO_GELU>
__global__ void cim_gemm_kernel(const signed char* __restrict__ Aq,
                                const signed char* __restrict__ Bq,
                                const float* __restrict__ corrA,
                                const float* __restrict__ corrB,
                                const float* __restrict__ bias,
                                const unsigned* __restrict__ slotA,
                                const unsigned* __restrict__ slotB,
                                int M, int N, int K, int nseg,
                                float* __restrict__ out,
                                unsigned* __restrict__ outAbsMax) {
    const int tid  = threadIdx.x;
    const int wave = tid >> 5;
    const int lane = tid & 31;
    const int half = lane >> 4;
    const int l15  = lane & 15;

    const int rowTile = blockIdx.y * 16;
    const int colTile = blockIdx.x * 128 + wave * 32;

    const signed char* aptr  = Aq + (size_t)(rowTile + l15) * K + half * 16;
    const signed char* bptr0 = Bq + (size_t)(colTile + l15) * K + half * 16;
    const signed char* bptr1 = bptr0 + (size_t)16 * K;

    v8i c0 = {};
    v8i c1 = {};

    v8i a_cur  = load_a_frag(aptr);
    v8i b0_cur = load_b_frag(bptr0);
    v8i b1_cur = load_b_frag(bptr1);

    for (int kb = 0; kb < K - 64; kb += 64) {
        const int kn = kb + 64;
        v8i a_n  = load_a_frag(aptr  + kn);
        v8i b0_n = load_b_frag(bptr0 + kn);
        v8i b1_n = load_b_frag(bptr1 + kn);
        __builtin_prefetch(aptr  + kn + 64, 0, 1);
        __builtin_prefetch(bptr0 + kn + 64, 0, 1);
        __builtin_prefetch(bptr1 + kn + 64, 0, 1);

        c0 = __builtin_amdgcn_wmma_i32_16x16x64_iu8(true, a_cur, true, b0_cur,
                                                    c0, false, false);
        c1 = __builtin_amdgcn_wmma_i32_16x16x64_iu8(true, a_cur, true, b1_cur,
                                                    c1, false, false);
        a_cur = a_n; b0_cur = b0_n; b1_cur = b1_n;
    }
    c0 = __builtin_amdgcn_wmma_i32_16x16x64_iu8(true, a_cur, true, b0_cur,
                                                c0, false, false);
    c1 = __builtin_amdgcn_wmma_i32_16x16x64_iu8(true, a_cur, true, b1_cur,
                                                c1, false, false);

    const float sA    = fmaxf(__uint_as_float(*slotA) / 7.0f, 1e-8f);
    const float sB    = fmaxf(__uint_as_float(*slotB) / 7.0f, 1e-8f);
    const float scale = sA * sB;
    const int   col0  = colTile + l15;
    const int   col1  = col0 + 16;
    const float bc0   = bias[col0];
    const float bc1   = bias[col1];
    float lmax = 0.0f;
#pragma unroll
    for (int r = 0; r < 8; ++r) {
        const int row = rowTile + r + half * 8;
        float corr0 = 0.0f, corr1 = 0.0f;
        for (int s = 0; s < nseg; ++s) {
            const float ca = corrA[row * nseg + s];
            corr0 += ca * corrB[col0 * nseg + s];
            corr1 += ca * corrB[col1 * nseg + s];
        }
        float v0 = ((float)c0[r] - corr0) * scale + bc0;
        float v1 = ((float)c1[r] - corr1) * scale + bc1;
        if (DO_GELU) {
            v0 = 0.5f * v0 * (1.0f + erff(v0 * 0.70710678118654752f));
            v1 = 0.5f * v1 * (1.0f + erff(v1 * 0.70710678118654752f));
        }
        out[(size_t)row * N + col0] = v0;
        out[(size_t)row * N + col1] = v1;
        lmax = fmaxf(lmax, fmaxf(fabsf(v0), fabsf(v1)));
    }

    if (outAbsMax != nullptr) {
        __shared__ unsigned smax;
        if (tid == 0) smax = 0u;
        __syncthreads();
        atomicMax(&smax, __float_as_uint(lmax));
        __syncthreads();
        if (tid == 0) atomicMax(outAbsMax, smax);
    }
}

extern "C" void kernel_launch(void* const* d_in, const int* in_sizes, int n_in,
                              void* d_out, int out_size, void* d_ws, size_t ws_size,
                              hipStream_t stream) {
    (void)n_in; (void)out_size; (void)ws_size;
    const float* x  = (const float*)d_in[0];
    const float* w1 = (const float*)d_in[1];
    const float* b1 = (const float*)d_in[2];
    const float* w2 = (const float*)d_in[3];
    const float* b2 = (const float*)d_in[4];
    float*       out = (float*)d_out;

    const int D = in_sizes[4];
    const int H = in_sizes[2];
    const int M = in_sizes[0] / D;
    const int nsegD = D / 256;
    const int nsegH = H / 256;

    uint8_t* p = (uint8_t*)d_ws;
    auto carve = [&](size_t bytes) -> uint8_t* {
        uint8_t* r = p;
        p += (bytes + 255) & ~(size_t)255;
        return r;
    };
    unsigned*    slots = (unsigned*)carve(256);
    signed char* xq    = (signed char*)carve((size_t)M * D);
    signed char* w1q   = (signed char*)carve((size_t)H * D);
    signed char* w2q   = (signed char*)carve((size_t)D * H);
    signed char* hq    = (signed char*)carve((size_t)M * H);
    float*       h     = (float*)carve((size_t)M * H * sizeof(float));
    float*       cAx   = (float*)carve((size_t)M * nsegD * sizeof(float));
    float*       cBw1  = (float*)carve((size_t)H * nsegD * sizeof(float));
    float*       cAh   = (float*)carve((size_t)M * nsegH * sizeof(float));
    float*       cBw2  = (float*)carve((size_t)D * nsegH * sizeof(float));

    init_slots_kernel<<<1, 32, 0, stream>>>(slots);

    absmax_kernel<<<1024, 256, 0, stream>>>(x,  M * D, slots + 0);
    absmax_kernel<<<1024, 256, 0, stream>>>(w1, H * D, slots + 1);
    absmax_kernel<<<1024, 256, 0, stream>>>(w2, D * H, slots + 3);

    quant_pack_kernel<<<M, 256, 0, stream>>>(x,  D, nsegD, slots + 0, xq,  cAx);
    quant_pack_kernel<<<H, 256, 0, stream>>>(w1, D, nsegD, slots + 1, w1q, cBw1);
    quant_pack_kernel<<<D, 256, 0, stream>>>(w2, H, nsegH, slots + 3, w2q, cBw2);

    cim_gemm_kernel<true><<<dim3(H / 128, M / 16), 128, 0, stream>>>(
        xq, w1q, cAx, cBw1, b1, slots + 0, slots + 1,
        M, H, D, nsegD, h, slots + 2);

    quant_pack_kernel<<<M, 256, 0, stream>>>(h, H, nsegH, slots + 2, hq, cAh);

    cim_gemm_kernel<false><<<dim3(D / 128, M / 16), 128, 0, stream>>>(
        hq, w2q, cAh, cBw2, b2, slots + 2, slots + 3,
        M, D, H, nsegH, out, nullptr);
}
